// GraphNetBlock_8126078124038
// MI455X (gfx1250) — matrix operand order repaired and checked by reading, not compiled, not run on hardware
//
#include <hip/hip_runtime.h>
#include <hip/hip_bf16.h>
#include <stdint.h>

typedef __bf16 bf16_t;
typedef __attribute__((ext_vector_type(16))) __bf16 v16bf;
typedef __attribute__((ext_vector_type(8)))  float  v8f;
typedef __attribute__((ext_vector_type(4)))  float  v4f;
typedef __attribute__((ext_vector_type(4)))  unsigned int v4u;

#define NN 100000
#define NE 800000
#define DD 128
#define HS 136    // hidden / W2 LDS row stride (128 + 8 pad -> 4-bank skew)
#define W1E 392   // edge W1 LDS row stride (384 + 8)
#define W1N 264   // node W1 LDS row stride (256 + 8)

__device__ __forceinline__ bf16_t f2bf(float x) { return (bf16_t)x; }

__device__ __forceinline__ void load_chunk8(const float* __restrict__ p, bf16_t* dst) {
    v4f a = *(const v4f*)p;
    v4f b = *(const v4f*)(p + 4);
    dst[0] = f2bf(a.x); dst[1] = f2bf(a.y); dst[2] = f2bf(a.z); dst[3] = f2bf(a.w);
    dst[4] = f2bf(b.x); dst[5] = f2bf(b.y); dst[6] = f2bf(b.z); dst[7] = f2bf(b.w);
}

// A fragment from global f32 row: contiguous [Kseg+16*hi, +16)
__device__ __forceinline__ v16bf loadA_g(const float* __restrict__ rb, int Kseg, int hi) {
    union { v16bf v; bf16_t e8[16]; } A;
    load_chunk8(rb + Kseg + hi * 16,     &A.e8[0]);
    load_chunk8(rb + Kseg + hi * 16 + 8, &A.e8[8]);
    return A.v;
}

// A fragment from LDS bf16 (contiguous 16 elements = 32B)
__device__ __forceinline__ v16bf loadA_lds(const bf16_t* p) {
    union { v16bf v; v4u q[2]; } A;
    A.q[0] = *(const v4u*)(p);
    A.q[1] = *(const v4u*)(p + 8);
    return A.v;
}

// B fragment from LDS bf16 (contiguous 16 elements = 32B)
__device__ __forceinline__ v16bf loadB_lds(const bf16_t* p) {
    union { v16bf v; v4u q[2]; } B;
    B.q[0] = *(const v4u*)(p);
    B.q[1] = *(const v4u*)(p + 8);
    return B.v;
}

__device__ __forceinline__ void atomic_add_f32(float* p, float v) {
    __hip_atomic_fetch_add(p, v, __ATOMIC_RELAXED, __HIP_MEMORY_SCOPE_AGENT);
}

// copy one weight matrix (bf16, row length K) into padded LDS (stride Kp), 16B chunks
__device__ __forceinline__ void stage_weights(const bf16_t* __restrict__ g, bf16_t* l,
                                              int K, int Kp, int rows, int tid, int nthreads) {
    const int cpr = K / 8;
    for (int c = tid; c < rows * cpr; c += nthreads) {
        const int row = c / cpr, kc = (c - row * cpr) * 8;
        *(v4u*)(&l[row * Kp + kc]) = *(const v4u*)(&g[row * K + kc]);
    }
}

__global__ void zero_f32_kernel(float* __restrict__ p, int n4) {
    int i = blockIdx.x * blockDim.x + threadIdx.x;
    if (i < n4) ((v4f*)p)[i] = (v4f){0.f, 0.f, 0.f, 0.f};
}

// transpose + convert: W[K][N] f32 row-major -> Wt[N][K] bf16
__global__ void prep_wt_kernel(const float* __restrict__ W, bf16_t* __restrict__ Wt,
                               int K, int Nout) {
    int idx = blockIdx.x * blockDim.x + threadIdx.x;
    if (idx < K * Nout) {
        int n = idx / K, k = idx % K;
        Wt[idx] = f2bf(W[(size_t)k * Nout + n]);
    }
}

// ============================ EDGE MLP ============================
__launch_bounds__(128)
__global__ void edge_mlp_kernel(const float* __restrict__ nf, const float* __restrict__ ef,
                                const int* __restrict__ senders, const int* __restrict__ receivers,
                                const bf16_t* __restrict__ W1t, const float* __restrict__ b1,
                                const bf16_t* __restrict__ W2t, const float* __restrict__ b2,
                                float* __restrict__ out_edges, float* __restrict__ agg) {
    __shared__ bf16_t lds_w1[128 * W1E];       // 100352 B
    __shared__ bf16_t lds_w2[128 * HS];        //  34816 B
    __shared__ bf16_t lds_h[4][16 * HS];       //  17408 B
    const int wave = threadIdx.x >> 5;
    const int lane = threadIdx.x & 31;
    const int r    = lane & 15;
    const int hi   = lane >> 4;

    stage_weights(W1t, lds_w1, 384, W1E, 128, threadIdx.x, blockDim.x);
    stage_weights(W2t, lds_w2, 128, HS,  128, threadIdx.x, blockDim.x);
    __syncthreads();

    float bias1[8], bias2[8];
    #pragma unroll
    for (int nt = 0; nt < 8; ++nt) { bias1[nt] = b1[nt * 16 + r]; bias2[nt] = b2[nt * 16 + r]; }

    const int ntiles = NE / 16;
    for (int tile = blockIdx.x * 4 + wave; tile < ntiles; tile += gridDim.x * 4) {
        asm volatile("" ::: "memory");   // keep weight ds_loads inside the loop (no LICM+spill)
        const int base = tile * 16;
        const int e  = base + r;
        const int sA = senders[e];
        const int rA = receivers[e];
        const float* rowbase[3] = { nf + (size_t)sA * DD,
                                    nf + (size_t)rA * DD,
                                    ef + (size_t)e  * DD };
        // ---- layer 1: [16x384] x [384x128], software-pipelined A and B ----
        v8f acc[8] = {};
        v16bf Acur = loadA_g(rowbase[0], 0, hi), Anext;
        v16bf Bcur = loadB_lds(&lds_w1[r * W1E + hi * 16]), Bnext;
        #pragma unroll
        for (int ks = 0; ks < 12; ++ks) {
            if (ks < 11) Anext = loadA_g(rowbase[(ks + 1) >> 2], ((ks + 1) & 3) * 32, hi);
            const int Kb = ks * 32;
            #pragma unroll
            for (int nt = 0; nt < 8; ++nt) {
                if (!(ks == 11 && nt == 7)) {
                    const int nks = (nt == 7) ? ks + 1 : ks;
                    const int nnt = (nt == 7) ? 0 : nt + 1;
                    Bnext = loadB_lds(&lds_w1[(nnt * 16 + r) * W1E + nks * 32 + hi * 16]);
                }
                acc[nt] = __builtin_amdgcn_wmma_f32_16x16x32_bf16(
                    false, Acur, false, Bcur, (short)0, acc[nt], false, false);
                Bcur = Bnext;
            }
            Acur = Anext;
        }
        // ---- bias + relu -> LDS bf16 (C layout -> A layout re-swizzle) ----
        #pragma unroll
        for (int nt = 0; nt < 8; ++nt) {
            const int col = nt * 16 + r;
            #pragma unroll
            for (int v = 0; v < 8; ++v) {
                float x = acc[nt][v] + bias1[nt];
                x = x > 0.f ? x : 0.f;
                lds_h[wave][(v + 8 * hi) * HS + col] = f2bf(x);
            }
        }
        // ---- layer 2: [16x128] x [128x128], software-pipelined ----
        v8f acc2[8] = {};
        Acur = loadA_lds(&lds_h[wave][r * HS + hi * 16]);
        Bcur = loadB_lds(&lds_w2[r * HS + hi * 16]);
        #pragma unroll
        for (int ks = 0; ks < 4; ++ks) {
            if (ks < 3) Anext = loadA_lds(&lds_h[wave][r * HS + (ks + 1) * 32 + hi * 16]);
            #pragma unroll
            for (int nt = 0; nt < 8; ++nt) {
                if (!(ks == 3 && nt == 7)) {
                    const int nks = (nt == 7) ? ks + 1 : ks;
                    const int nnt = (nt == 7) ? 0 : nt + 1;
                    Bnext = loadB_lds(&lds_w2[(nnt * 16 + r) * HS + nks * 32 + hi * 16]);
                }
                acc2[nt] = __builtin_amdgcn_wmma_f32_16x16x32_bf16(
                    false, Acur, false, Bcur, (short)0, acc2[nt], false, false);
                Bcur = Bnext;
            }
            Acur = Anext;
        }
        // ---- epilogue: new_edges = ef + e_upd ; agg[recv] += e_upd ----
        #pragma unroll
        for (int v = 0; v < 8; ++v) {
            const int m  = v + 8 * hi;
            const int em = base + m;
            const int rm = receivers[em];
            const float* efr = ef + (size_t)em * DD;
            float* oer = out_edges + (size_t)em * DD;
            float* agr = agg + (size_t)rm * DD;
            #pragma unroll
            for (int nt = 0; nt < 8; ++nt) {
                const int col = nt * 16 + r;
                const float val = acc2[nt][v] + bias2[nt];
                oer[col] = efr[col] + val;
                atomic_add_f32(&agr[col], val);
            }
        }
    }
}

// ============================ NODE MLP ============================
// nodebuf holds agg on entry, new_nodes on exit (in-place, race-free per tile).
__launch_bounds__(128)
__global__ void node_mlp_kernel(const float* __restrict__ nf,
                                const bf16_t* __restrict__ W1t, const float* __restrict__ b1,
                                const bf16_t* __restrict__ W2t, const float* __restrict__ b2,
                                float* __restrict__ nodebuf) {
    __shared__ bf16_t lds_w1[128 * W1N];       // 67584 B
    __shared__ bf16_t lds_w2[128 * HS];        // 34816 B
    __shared__ bf16_t lds_h[4][16 * HS];       // 17408 B
    const int wave = threadIdx.x >> 5;
    const int lane = threadIdx.x & 31;
    const int r    = lane & 15;
    const int hi   = lane >> 4;

    stage_weights(W1t, lds_w1, 256, W1N, 128, threadIdx.x, blockDim.x);
    stage_weights(W2t, lds_w2, 128, HS,  128, threadIdx.x, blockDim.x);
    __syncthreads();

    float bias1[8], bias2[8];
    #pragma unroll
    for (int nt = 0; nt < 8; ++nt) { bias1[nt] = b1[nt * 16 + r]; bias2[nt] = b2[nt * 16 + r]; }

    const int ntiles = NN / 16;
    for (int tile = blockIdx.x * 4 + wave; tile < ntiles; tile += gridDim.x * 4) {
        asm volatile("" ::: "memory");
        const int base = tile * 16;
        const int node = base + r;
        const float* rowbase[2] = { nf + (size_t)node * DD,
                                    nodebuf + (size_t)node * DD };
        // ---- layer 1: [16x256] x [256x128] ----
        v8f acc[8] = {};
        v16bf Acur = loadA_g(rowbase[0], 0, hi), Anext;
        v16bf Bcur = loadB_lds(&lds_w1[r * W1N + hi * 16]), Bnext;
        #pragma unroll
        for (int ks = 0; ks < 8; ++ks) {
            if (ks < 7) Anext = loadA_g(rowbase[(ks + 1) >> 2], ((ks + 1) & 3) * 32, hi);
            #pragma unroll
            for (int nt = 0; nt < 8; ++nt) {
                if (!(ks == 7 && nt == 7)) {
                    const int nks = (nt == 7) ? ks + 1 : ks;
                    const int nnt = (nt == 7) ? 0 : nt + 1;
                    Bnext = loadB_lds(&lds_w1[(nnt * 16 + r) * W1N + nks * 32 + hi * 16]);
                }
                acc[nt] = __builtin_amdgcn_wmma_f32_16x16x32_bf16(
                    false, Acur, false, Bcur, (short)0, acc[nt], false, false);
                Bcur = Bnext;
            }
            Acur = Anext;
        }
        // ---- bias + relu -> LDS ----
        #pragma unroll
        for (int nt = 0; nt < 8; ++nt) {
            const int col = nt * 16 + r;
            #pragma unroll
            for (int v = 0; v < 8; ++v) {
                float x = acc[nt][v] + bias1[nt];
                x = x > 0.f ? x : 0.f;
                lds_h[wave][(v + 8 * hi) * HS + col] = f2bf(x);
            }
        }
        // ---- layer 2 ----
        v8f acc2[8] = {};
        Acur = loadA_lds(&lds_h[wave][r * HS + hi * 16]);
        Bcur = loadB_lds(&lds_w2[r * HS + hi * 16]);
        #pragma unroll
        for (int ks = 0; ks < 4; ++ks) {
            if (ks < 3) Anext = loadA_lds(&lds_h[wave][r * HS + (ks + 1) * 32 + hi * 16]);
            #pragma unroll
            for (int nt = 0; nt < 8; ++nt) {
                if (!(ks == 3 && nt == 7)) {
                    const int nks = (nt == 7) ? ks + 1 : ks;
                    const int nnt = (nt == 7) ? 0 : nt + 1;
                    Bnext = loadB_lds(&lds_w2[(nnt * 16 + r) * HS + nks * 32 + hi * 16]);
                }
                acc2[nt] = __builtin_amdgcn_wmma_f32_16x16x32_bf16(
                    false, Acur, false, Bcur, (short)0, acc2[nt], false, false);
                Bcur = Bnext;
            }
            Acur = Anext;
        }
        // ---- epilogue: new_nodes = nf + n_upd (overwrites agg rows) ----
        #pragma unroll
        for (int v = 0; v < 8; ++v) {
            const int m  = v + 8 * hi;
            const int nm = base + m;
            const float* nfr = nf + (size_t)nm * DD;
            float* onr = nodebuf + (size_t)nm * DD;
            #pragma unroll
            for (int nt = 0; nt < 8; ++nt) {
                const int col = nt * 16 + r;
                onr[col] = nfr[col] + acc2[nt][v] + bias2[nt];
            }
        }
    }
}

extern "C" void kernel_launch(void* const* d_in, const int* in_sizes, int n_in,
                              void* d_out, int out_size, void* d_ws, size_t ws_size,
                              hipStream_t stream) {
    const float* nf  = (const float*)d_in[0];
    const float* ef  = (const float*)d_in[1];
    const int* senders   = (const int*)d_in[2];
    const int* receivers = (const int*)d_in[3];
    const float* We1 = (const float*)d_in[4];
    const float* be1 = (const float*)d_in[5];
    const float* We2 = (const float*)d_in[6];
    const float* be2 = (const float*)d_in[7];
    const float* Wn1 = (const float*)d_in[8];
    const float* bn1 = (const float*)d_in[9];
    const float* Wn2 = (const float*)d_in[10];
    const float* bn2 = (const float*)d_in[11];

    float* out       = (float*)d_out;
    float* out_nodes = out;                         // doubles as agg scratch
    float* out_edges = out + (size_t)NN * DD;

    char* ws = (char*)d_ws;                         // bf16 transposed weights (~230 KB)
    bf16_t* We1t = (bf16_t*)(ws + 0);               // [128][384]
    bf16_t* We2t = (bf16_t*)(ws + 98304);           // [128][128]
    bf16_t* Wn1t = (bf16_t*)(ws + 131072);          // [128][256]
    bf16_t* Wn2t = (bf16_t*)(ws + 196608);          // [128][128]

    const int n4 = NN * DD / 4;
    zero_f32_kernel<<<(n4 + 255) / 256, 256, 0, stream>>>(out_nodes, n4);
    prep_wt_kernel<<<(384 * 128 + 255) / 256, 256, 0, stream>>>(We1, We1t, 384, 128);
    prep_wt_kernel<<<(128 * 128 + 255) / 256, 256, 0, stream>>>(We2, We2t, 128, 128);
    prep_wt_kernel<<<(256 * 128 + 255) / 256, 256, 0, stream>>>(Wn1, Wn1t, 256, 128);
    prep_wt_kernel<<<(128 * 128 + 255) / 256, 256, 0, stream>>>(Wn2, Wn2t, 128, 128);

    edge_mlp_kernel<<<1024, 128, 0, stream>>>(
        nf, ef, senders, receivers, We1t, be1, We2t, be2, out_edges, out_nodes);
    node_mlp_kernel<<<512, 128, 0, stream>>>(
        nf, Wn1t, bn1, Wn2t, bn2, out_nodes);
}
